// MLA_67379446940108
// MI455X (gfx1250) — matrix operand order repaired and checked by reading, not compiled, not run on hardware
//
#include <hip/hip_runtime.h>
#include <math.h>

// ---------------- constants ----------------
#define Tt   2048
#define Cc   2048
#define Hh   16
#define DN   128
#define DR   64
#define DV   128
#define QKD  192      // DN + DR
#define RQ   1536
#define RKV  512
#define EPSc 1e-6f
#define SCALEc 0.07216878364870323f   // 192^-0.5

typedef __bf16 bf16_t;
typedef __attribute__((ext_vector_type(16))) __bf16 v16bf;
typedef __attribute__((ext_vector_type(8)))  __bf16 v8bf;
typedef __attribute__((ext_vector_type(8)))  float  v8f;

// ---------------- WMMA helpers ----------------
__device__ __forceinline__ v8f wmma_bf16(v16bf a, v16bf b, v8f c) {
    // D = A(16x32 bf16) x B(32x16 bf16) + C(16x16 f32)
    return __builtin_amdgcn_wmma_f32_16x16x32_bf16(
        /*neg_a=*/false, a, /*neg_b=*/false, b,
        /*c_mod=*/(short)0, c, /*reuse_a=*/false, /*reuse_b=*/false);
}

// A-fragment (16x32, row-major source). lane row = base row, half = lane>>4.
// elements e0..7  -> K = k0 + 16*half + e      (contiguous 8 bf16)
// elements e8..15 -> K = k0 + 16*half + e      (contiguous 8 bf16)
__device__ __forceinline__ v16bf load_a_frag(const bf16_t* __restrict__ rowk0, int half) {
    v8bf lo = *(const v8bf*)(rowk0 + 16 * half);
    v8bf hi = *(const v8bf*)(rowk0 + 16 * half + 8);
    v16bf r;
#pragma unroll
    for (int i = 0; i < 8; ++i) { r[i] = lo[i]; r[i + 8] = hi[i]; }
    return r;
}

// B-fragment (32x16): lane column = n, elements = contiguous K (16 bf16 = 32B).
__device__ __forceinline__ v16bf load_b_frag(const bf16_t* __restrict__ p) {
    return *(const v16bf*)p;
}

// ---------------- elementwise convert ----------------
__global__ void k_f32_to_bf16(const float* __restrict__ in, bf16_t* __restrict__ out, size_t n) {
    size_t i = (size_t)blockIdx.x * blockDim.x + threadIdx.x;
    if (i < n) out[i] = (bf16_t)in[i];
}

// ---------------- generic WMMA GEMM: C[M,N] = A[M,K] * W[N,K]^T + bias ----------------
// 128 threads = 4 waves (2x2), wave tile 64x64 -> block tile 128x128.
// 16 WMMAs per k-step against 8 A/B fragments: 1.0 b128-loads per WMMA.
// N may be ragged (guarded stores); W staging buffer is padded so OOB rows are readable.
__global__ __launch_bounds__(128) void k_gemm_bf16_nt(
    const bf16_t* __restrict__ A, const bf16_t* __restrict__ W,
    const float* __restrict__ bias, float* __restrict__ Cout,
    int M, int N, int K)
{
    const int lane = threadIdx.x & 31;
    const int wave = threadIdx.x >> 5;
    const int waveM = wave & 1, waveN = wave >> 1;   // 2x2 wave grid
    const int m0 = blockIdx.x * 128 + waveM * 64;
    const int n0 = blockIdx.y * 128 + waveN * 64;
    const int half = lane >> 4, l16 = lane & 15;

    v8f acc[4][4] = {};
    const bf16_t* arow[4];
    const bf16_t* brow[4];
#pragma unroll
    for (int i = 0; i < 4; ++i) {
        arow[i] = A + (size_t)(m0 + i * 16 + l16) * K;
        brow[i] = W + (size_t)(n0 + i * 16 + l16) * K + 16 * half;
    }

    for (int k0 = 0; k0 < K; k0 += 32) {
        v16bf a[4], b[4];
#pragma unroll
        for (int i = 0; i < 4; ++i) a[i] = load_a_frag(arow[i] + k0, half);
#pragma unroll
        for (int i = 0; i < 4; ++i) b[i] = load_b_frag(brow[i] + k0);
#pragma unroll
        for (int mt = 0; mt < 4; ++mt)
#pragma unroll
            for (int nt = 0; nt < 4; ++nt)
                acc[mt][nt] = wmma_bf16(a[mt], b[nt], acc[mt][nt]);
    }

#pragma unroll
    for (int mt = 0; mt < 4; ++mt)
#pragma unroll
        for (int nt = 0; nt < 4; ++nt) {
            int n = n0 + nt * 16 + l16;
            if (n < N) {
                float bv = bias ? bias[n] : 0.0f;
#pragma unroll
                for (int r = 0; r < 8; ++r) {
                    int m = m0 + mt * 16 + r + 8 * half;
                    Cout[(size_t)m * N + n] = acc[mt][nt][r] + bv;
                }
            }
        }
}

// ---------------- RMSNorm (f32 in, bf16 out) ----------------
__global__ __launch_bounds__(256) void k_rmsnorm_bf16(
    const float* __restrict__ in, int in_stride,
    const float* __restrict__ w,
    bf16_t* __restrict__ out, int out_stride, int D)
{
    const int t = blockIdx.x;
    const float* row = in + (size_t)t * in_stride;
    __shared__ float red[256];
    float s = 0.0f;
    for (int i = threadIdx.x; i < D; i += 256) { float v = row[i]; s += v * v; }
    red[threadIdx.x] = s;
    __syncthreads();
    for (int o = 128; o > 0; o >>= 1) {
        if ((int)threadIdx.x < o) red[threadIdx.x] += red[threadIdx.x + o];
        __syncthreads();
    }
    float scale = rsqrtf(red[0] / (float)D + EPSc);
    for (int i = threadIdx.x; i < D; i += 256)
        out[(size_t)t * out_stride + i] = (bf16_t)(row[i] * scale * w[i]);
}

// ---------------- assemble Q: [T,H*QKD] f32 -> Qb[h][t][QKD] bf16, rope on q_pe ----------------
__global__ __launch_bounds__(256) void k_assemble_q(
    const float* __restrict__ qf, const float* __restrict__ freqs,
    bf16_t* __restrict__ Qb)
{
    const int t = blockIdx.x;
    const float* qr = qf + (size_t)t * (Hh * QKD);
    for (int i = threadIdx.x; i < Hh * DN; i += 256) {
        int h = i / DN, d = i % DN;
        Qb[((size_t)h * Tt + t) * QKD + d] = (bf16_t)qr[h * QKD + d];
    }
    for (int i = threadIdx.x; i < Hh * (DR / 2); i += 256) {
        int h = i / (DR / 2), p = i % (DR / 2);
        float f = freqs[(size_t)t * (DR / 2) + p];
        float cs = cosf(f), sn = sinf(f);
        float xr = qr[h * QKD + DN + 2 * p];
        float xi = qr[h * QKD + DN + 2 * p + 1];
        Qb[((size_t)h * Tt + t) * QKD + DN + 2 * p]     = (bf16_t)(xr * cs - xi * sn);
        Qb[((size_t)h * Tt + t) * QKD + DN + 2 * p + 1] = (bf16_t)(xr * sn + xi * cs);
    }
}

// ---------------- assemble K: k_nope from kvb, roped k_pe broadcast -> Kb[h][t][QKD] ----------------
__global__ __launch_bounds__(256) void k_assemble_k(
    const float* __restrict__ kvb, const float* __restrict__ kvpe,
    const float* __restrict__ freqs, bf16_t* __restrict__ Kb)
{
    const int t = blockIdx.x;
    const float* kr = kvb + (size_t)t * (Hh * (DN + DV));
    for (int i = threadIdx.x; i < Hh * DN; i += 256) {
        int h = i / DN, d = i % DN;
        Kb[((size_t)h * Tt + t) * QKD + d] = (bf16_t)kr[h * (DN + DV) + d];
    }
    __shared__ float pe[DR];
    if ((int)threadIdx.x < DR / 2) {
        int p = threadIdx.x;
        float f = freqs[(size_t)t * (DR / 2) + p];
        float cs = cosf(f), sn = sinf(f);
        float xr = kvpe[(size_t)t * (RKV + DR) + RKV + 2 * p];
        float xi = kvpe[(size_t)t * (RKV + DR) + RKV + 2 * p + 1];
        pe[2 * p]     = xr * cs - xi * sn;
        pe[2 * p + 1] = xr * sn + xi * cs;
    }
    __syncthreads();
    for (int i = threadIdx.x; i < Hh * DR; i += 256) {
        int h = i / DR, d = i % DR;
        Kb[((size_t)h * Tt + t) * QKD + DN + d] = (bf16_t)pe[d];
    }
}

// ---------------- assemble V transposed: Vt[h][d][t] bf16 ----------------
__global__ void k_assemble_v(const float* __restrict__ kvb, bf16_t* __restrict__ Vt) {
    size_t idx = (size_t)blockIdx.x * blockDim.x + threadIdx.x;
    size_t n = (size_t)Hh * DV * Tt;
    if (idx >= n) return;
    int t = (int)(idx % Tt);
    int rem = (int)(idx / Tt);
    int d = rem % DV;
    int h = rem / DV;
    Vt[idx] = (bf16_t)kvb[(size_t)t * (Hh * (DN + DV)) + h * (DN + DV) + DN + d];
}

// ---------------- flash attention: one wave = 16 query rows x 1 head ----------------
__global__ __launch_bounds__(32) void k_attn(
    const bf16_t* __restrict__ Qb, const bf16_t* __restrict__ Kb,
    const bf16_t* __restrict__ Vt, float* __restrict__ Y)
{
    const int h = blockIdx.y;
    const int qb = blockIdx.x * 16;
    const int lane = threadIdx.x & 31;
    const int half = lane >> 4, l16 = lane & 15;

    alignas(32) __shared__ bf16_t Pl[16 * 32];

    const bf16_t* Qh = Qb + (size_t)h * Tt * QKD;
    const bf16_t* Kh = Kb + (size_t)h * Tt * QKD;
    const bf16_t* Vh = Vt + (size_t)h * DV * Tt;

    v8f accO[8];
#pragma unroll
    for (int i = 0; i < 8; ++i) accO[i] = (v8f)0.0f;
    float mrow[8], lrow[8];
#pragma unroll
    for (int r = 0; r < 8; ++r) { mrow[r] = -INFINITY; lrow[r] = 0.0f; }

    const bf16_t* qrow = Qh + (size_t)(qb + l16) * QKD;

    const int kend = ((qb + 16 + 31) >> 5) << 5;   // key blocks covering keys <= qb+15
    for (int kb = 0; kb < kend; kb += 32) {
        // ---- S = Q K^T over QKD=192 (6 k-steps), 2 key sub-tiles ----
        v8f sacc[2] = {};
        const bf16_t* krow0 = Kh + (size_t)(kb + l16) * QKD + 16 * half;
        const bf16_t* krow1 = Kh + (size_t)(kb + 16 + l16) * QKD + 16 * half;
#pragma unroll
        for (int ks = 0; ks < QKD; ks += 32) {
            v16bf a  = load_a_frag(qrow + ks, half);
            v16bf b0 = load_b_frag(krow0 + ks);
            v16bf b1 = load_b_frag(krow1 + ks);
            sacc[0] = wmma_bf16(a, b0, sacc[0]);
            sacc[1] = wmma_bf16(a, b1, sacc[1]);
        }
        // ---- scale (+ causal mask on diagonal-overlapping blocks) ----
        if (kb + 31 > qb) {
#pragma unroll
            for (int nt = 0; nt < 2; ++nt)
#pragma unroll
                for (int r = 0; r < 8; ++r) {
                    int key = kb + nt * 16 + l16;
                    int row = qb + r + 8 * half;
                    float v = sacc[nt][r] * SCALEc;
                    sacc[nt][r] = (key > row) ? -INFINITY : v;
                }
        } else {
#pragma unroll
            for (int nt = 0; nt < 2; ++nt)
#pragma unroll
                for (int r = 0; r < 8; ++r) sacc[nt][r] *= SCALEc;
        }
        // ---- online softmax: row max / sum via width-16 xor shuffles ----
        float alpha[8];
#pragma unroll
        for (int r = 0; r < 8; ++r) {
            float v = fmaxf(sacc[0][r], sacc[1][r]);
#pragma unroll
            for (int o = 8; o > 0; o >>= 1) v = fmaxf(v, __shfl_xor(v, o, 16));
            float mn = fmaxf(mrow[r], v);
            alpha[r] = __expf(mrow[r] - mn);
            mrow[r] = mn;
        }
#pragma unroll
        for (int r = 0; r < 8; ++r) {
            float p0 = __expf(sacc[0][r] - mrow[r]);
            float p1 = __expf(sacc[1][r] - mrow[r]);
            sacc[0][r] = p0; sacc[1][r] = p1;
            float v = p0 + p1;
#pragma unroll
            for (int o = 8; o > 0; o >>= 1) v += __shfl_xor(v, o, 16);
            lrow[r] = lrow[r] * alpha[r] + v;
        }
#pragma unroll
        for (int dvt = 0; dvt < 8; ++dvt)
#pragma unroll
            for (int r = 0; r < 8; ++r) accO[dvt][r] *= alpha[r];

        // ---- P (C-layout) -> LDS row-major [16][32] -> A-fragment ----
#pragma unroll
        for (int nt = 0; nt < 2; ++nt)
#pragma unroll
            for (int r = 0; r < 8; ++r)
                Pl[(r + 8 * half) * 32 + nt * 16 + l16] = (bf16_t)sacc[nt][r];
        __syncthreads();
        v16bf ap;
        {
            v8bf lo = *(const v8bf*)(&Pl[l16 * 32 + 16 * half]);
            v8bf hi = *(const v8bf*)(&Pl[l16 * 32 + 16 * half + 8]);
#pragma unroll
            for (int i = 0; i < 8; ++i) { ap[i] = lo[i]; ap[i + 8] = hi[i]; }
        }
        // ---- O += P V  (V pre-transposed: contiguous B-frag loads) ----
#pragma unroll
        for (int dvt = 0; dvt < 8; ++dvt) {
            v16bf bv = load_b_frag(Vh + (size_t)(dvt * 16 + l16) * Tt + kb + 16 * half);
            accO[dvt] = wmma_bf16(ap, bv, accO[dvt]);
        }
        __syncthreads();
    }

    // ---- epilogue: normalize, write y[t][h*DV + d] ----
#pragma unroll
    for (int dvt = 0; dvt < 8; ++dvt)
#pragma unroll
        for (int r = 0; r < 8; ++r) {
            int row = qb + r + 8 * half;
            Y[(size_t)row * Cc + h * DV + dvt * 16 + l16] = accO[dvt][r] / lrow[r];
        }
}

// ---------------- host side ----------------
extern "C" void kernel_launch(void* const* d_in, const int* in_sizes, int n_in,
                              void* d_out, int out_size, void* d_ws, size_t ws_size,
                              hipStream_t stream) {
    const float* x        = (const float*)d_in[0];
    const float* freqs    = (const float*)d_in[1];
    const float* wq_a     = (const float*)d_in[2];
    const float* bq_a     = (const float*)d_in[3];
    const float* q_norm_w = (const float*)d_in[4];
    const float* wq_b     = (const float*)d_in[5];
    const float* bq_b     = (const float*)d_in[6];
    const float* wkv_a    = (const float*)d_in[7];
    const float* bkv_a    = (const float*)d_in[8];
    const float* kv_norm_w= (const float*)d_in[9];
    const float* wkv_b    = (const float*)d_in[10];
    const float* bkv_b    = (const float*)d_in[11];
    const float* wo       = (const float*)d_in[12];
    const float* bo       = (const float*)d_in[13];
    float* out = (float*)d_out;

    char* ws = (char*)d_ws;
    // bump layout with lifetime-safe aliasing (bytes)
    const size_t OFF_XB     = 0;                                   // x bf16: 8,388,608
    const size_t OFF_WB     = OFF_XB + (size_t)Tt * Cc * 2;        // shared weight staging: 9,437,184
    const size_t OFF_R1     = OFF_WB + (size_t)3072 * 1536 * 2;    // tq | kvb | y : 33,554,432
    const size_t OFF_QF     = OFF_R1 + (size_t)Tt * 4096 * 4;      // qf | Vt : 25,165,824
    const size_t OFF_QLATB  = OFF_QF + (size_t)Tt * 3072 * 4;      // 6,291,456
    const size_t OFF_KVPE   = OFF_QLATB + (size_t)Tt * RQ * 2;     // 4,718,592
    const size_t OFF_KVLATB = OFF_KVPE + (size_t)Tt * (RKV + DR) * 4; // 2,097,152
    const size_t OFF_QB     = OFF_KVLATB + (size_t)Tt * RKV * 2;   // Qb | yb : 12,582,912
    const size_t OFF_KB     = OFF_QB + (size_t)Hh * Tt * QKD * 2;  // 12,582,912

    bf16_t* xb     = (bf16_t*)(ws + OFF_XB);
    bf16_t* WB     = (bf16_t*)(ws + OFF_WB);
    float*  tq     = (float*) (ws + OFF_R1);
    float*  kvb    = (float*) (ws + OFF_R1);
    float*  y      = (float*) (ws + OFF_R1);
    float*  qf     = (float*) (ws + OFF_QF);
    bf16_t* Vt     = (bf16_t*)(ws + OFF_QF);
    bf16_t* qlatb  = (bf16_t*)(ws + OFF_QLATB);
    float*  kvpe   = (float*) (ws + OFF_KVPE);
    bf16_t* kvlatb = (bf16_t*)(ws + OFF_KVLATB);
    bf16_t* Qb     = (bf16_t*)(ws + OFF_QB);
    bf16_t* yb     = (bf16_t*)(ws + OFF_QB);
    bf16_t* Kb     = (bf16_t*)(ws + OFF_KB);

    auto cvt = [&](const float* src, bf16_t* dst, size_t n) {
        k_f32_to_bf16<<<dim3((unsigned)((n + 255) / 256)), dim3(256), 0, stream>>>(src, dst, n);
    };
    auto gemm = [&](const bf16_t* A, const bf16_t* W, const float* bias, float* Cout,
                    int M, int N, int K) {
        k_gemm_bf16_nt<<<dim3(M / 128, (N + 127) / 128), dim3(128), 0, stream>>>(A, W, bias, Cout, M, N, K);
    };

    // 0) x -> bf16
    cvt(x, xb, (size_t)Tt * Cc);

    // 1) q_lat_raw = x @ wq_a^T + bq_a   [T, RQ]
    cvt(wq_a, WB, (size_t)RQ * Cc);
    gemm(xb, WB, bq_a, tq, Tt, RQ, Cc);

    // 2) rmsnorm -> qlatb (bf16)
    k_rmsnorm_bf16<<<dim3(Tt), dim3(256), 0, stream>>>(tq, RQ, q_norm_w, qlatb, RQ, RQ);

    // 3) q = q_lat @ wq_b^T + bq_b   [T, H*QKD]
    cvt(wq_b, WB, (size_t)(Hh * QKD) * RQ);
    gemm(qlatb, WB, bq_b, qf, Tt, Hh * QKD, RQ);

    // 4) kvpe = x @ wkv_a^T + bkv_a   [T, RKV+DR]  (N=576 ragged: guarded stores)
    cvt(wkv_a, WB, (size_t)(RKV + DR) * Cc);
    gemm(xb, WB, bkv_a, kvpe, Tt, RKV + DR, Cc);

    // 5) rmsnorm(kv_lat) -> kvlatb (bf16); k_pe stays f32 in kvpe tail
    k_rmsnorm_bf16<<<dim3(Tt), dim3(256), 0, stream>>>(kvpe, RKV + DR, kv_norm_w, kvlatb, RKV, RKV);

    // 6) kvb = kv_lat_n @ wkv_b^T + bkv_b   [T, H*(DN+DV)]
    cvt(wkv_b, WB, (size_t)(Hh * (DN + DV)) * RKV);
    gemm(kvlatb, WB, bkv_b, kvb, Tt, Hh * (DN + DV), RKV);

    // 7) assemble Q (rope q_pe), K (rope k_pe broadcast), V^T
    k_assemble_q<<<dim3(Tt), dim3(256), 0, stream>>>(qf, freqs, Qb);
    k_assemble_k<<<dim3(Tt), dim3(256), 0, stream>>>(kvb, kvpe, freqs, Kb);
    {
        size_t n = (size_t)Hh * DV * Tt;
        k_assemble_v<<<dim3((unsigned)((n + 255) / 256)), dim3(256), 0, stream>>>(kvb, Vt);
    }

    // 8) flash attention -> y [T, C]   (kvb dead; y aliases R1)
    k_attn<<<dim3(Tt / 16, Hh), dim3(32), 0, stream>>>(Qb, Kb, Vt, y);

    // 9) out = y @ wo^T + bo
    cvt(y, yb, (size_t)Tt * Cc);
    cvt(wo, WB, (size_t)Cc * Cc);
    gemm(yb, WB, bo, out, Tt, Cc, Cc);
}
